// MultiHeadAttenion_27470610825596
// MI455X (gfx1250) — compile-verified, matrix operand order checked by reading, not run on hardware
//
#include <hip/hip_runtime.h>
#include <hip/hip_bf16.h>
#include <math.h>
#include <stdint.h>

#define DEV __device__ __forceinline__

constexpr int Bc = 2, Tc = 2048, Ec = 1024, Hc = 16, DHc = 64;
constexpr int BT = Bc * Tc;  // 4096 tokens

typedef __attribute__((ext_vector_type(16))) __bf16 bf16x16;
typedef __attribute__((ext_vector_type(8)))  __bf16 bf16x8;
typedef __attribute__((ext_vector_type(8)))  float  f32x8;
typedef __attribute__((ext_vector_type(4)))  float  f32x4;

DEV bf16x16 cat16(bf16x8 lo, bf16x8 hi) {
  return __builtin_shufflevector(lo, hi, 0,1,2,3,4,5,6,7,8,9,10,11,12,13,14,15);
}

DEV f32x8 wmma_bf16(bf16x16 a, bf16x16 b, f32x8 c) {
  // D = A(16x32 bf16) * B(32x16 bf16) + C(16x16 f32)
  return __builtin_amdgcn_wmma_f32_16x16x32_bf16(false, a, false, b, (short)0, c,
                                                 false, false);
}

// 16-bit fragment loader from a row-major row pointer (32 K-contiguous elems).
// Per ISA layout: lanes 0-15 take K{0..7,16..23}; lanes 16-31 take K{8..15,24..31}.
DEV bf16x16 load_frag_row(const __bf16* rowp, int hi) {
  bf16x8 lo  = *(const bf16x8*)(rowp + hi * 8);
  bf16x8 hi8 = *(const bf16x8*)(rowp + 16 + hi * 8);
  return cat16(lo, hi8);
}

// CDNA5 async copy: 16 bytes global -> LDS, tracked by ASYNCcnt.
DEV void async_copy_b128(unsigned lds_off, const void* gptr) {
  asm volatile("global_load_async_to_lds_b128 %0, %1, off"
               :: "v"(lds_off), "v"((uint64_t)(uintptr_t)gptr) : "memory");
}

// ---------------------------------------------------------------------------
// Kernel 0a: embed f32 [BT,E] -> bf16 [BT,E]  (one-time, bandwidth trivial)
// ---------------------------------------------------------------------------
__global__ __launch_bounds__(256) void convert_embed_kernel(
    const float* __restrict__ e, __bf16* __restrict__ out)
{
  size_t i = ((size_t)blockIdx.x * 256 + threadIdx.x) * 8;
  f32x4 x0 = *(const f32x4*)(e + i);
  f32x4 x1 = *(const f32x4*)(e + i + 4);
  bf16x8 y;
  #pragma unroll
  for (int j = 0; j < 4; ++j) { y[j] = (__bf16)x0[j]; y[4 + j] = (__bf16)x1[j]; }
  *(bf16x8*)(out + i) = y;
}

// ---------------------------------------------------------------------------
// Kernel 0b: W[h][E][DH] f32 -> Wt[h][DH][E] bf16 (transpose so the GEMM's
// B^T tiles are contiguous and async-copyable). grid (E/32, H, 3), block 256.
// ---------------------------------------------------------------------------
__global__ __launch_bounds__(256) void convert_weights_kernel(
    const float* __restrict__ Wq, const float* __restrict__ Wk,
    const float* __restrict__ Wv,
    __bf16* __restrict__ Wtq, __bf16* __restrict__ Wtk, __bf16* __restrict__ Wtv)
{
  const float* W; __bf16* Wt;
  if (blockIdx.z == 0)      { W = Wq; Wt = Wtq; }
  else if (blockIdx.z == 1) { W = Wk; Wt = Wtk; }
  else                      { W = Wv; Wt = Wtv; }
  const int h  = blockIdx.y;
  const int d  = threadIdx.x & 63;
  const int eg = blockIdx.x * 4 + (threadIdx.x >> 6);
  const int e0 = eg * 8;
  const float* src = W + ((size_t)h * Ec + e0) * DHc + d;
  bf16x8 y;
  #pragma unroll
  for (int j = 0; j < 8; ++j) y[j] = (__bf16)src[(size_t)j * DHc];
  *(bf16x8*)&Wt[((size_t)h * DHc + d) * Ec + e0] = y;
}

// ---------------------------------------------------------------------------
// Kernel 1: QKV projection, double-buffered async-to-LDS staging.
// grid (BT/128, H, 3), block 256 (8 waves). Q pre-scaled by 1/sqrt(DH).
// V is written TRANSPOSED [B,H,DH,T] so flash PV fragments load contiguously.
// ---------------------------------------------------------------------------
__global__ __launch_bounds__(256) void qkv_proj_kernel(
    const __bf16* __restrict__ Ebf,
    const __bf16* __restrict__ Wtq, const float* __restrict__ bq,
    const __bf16* __restrict__ Wtk, const float* __restrict__ bk,
    const __bf16* __restrict__ Wtv, const float* __restrict__ bv,
    __bf16* __restrict__ Qw, __bf16* __restrict__ Kw, __bf16* __restrict__ Vw)
{
  __shared__ alignas(16) __bf16 Asl[2][128 * 40];  // 128 rows x 32 K, pad 40
  __shared__ alignas(16) __bf16 Bsl[2][64 * 40];   // 64 cols(dh) x 32 K, pad 40

  const int tid = threadIdx.x;
  const int m0  = blockIdx.x * 128;
  const int h   = blockIdx.y;

  const __bf16* Wt; const float* bias; __bf16* outp; float scale;
  if (blockIdx.z == 0)      { Wt = Wtq; bias = bq; outp = Qw; scale = 0.125f; }
  else if (blockIdx.z == 1) { Wt = Wtk; bias = bk; outp = Kw; scale = 1.0f; }
  else                      { Wt = Wtv; bias = bv; outp = Vw; scale = 1.0f; }

  const __bf16* Ebase = Ebf + (size_t)m0 * Ec;
  const __bf16* Wbase = Wt + (size_t)h * DHc * Ec;

  const int lane = tid & 31, wave = tid >> 5;
  const int l = lane & 15, hi = lane >> 4;

  // each thread issues 3 async b128 copies per tile (2 for A, 1 for B)
  auto stage_tile = [&](int k0, int buf) {
    #pragma unroll
    for (int i = 0; i < 2; ++i) {
      int idx = tid + i * 256;     // 0..511
      int row = idx >> 2, g = idx & 3;
      async_copy_b128((unsigned)(uintptr_t)&Asl[buf][row * 40 + g * 8],
                      Ebase + (size_t)row * Ec + k0 + g * 8);
    }
    int d = tid >> 2, g = tid & 3;
    async_copy_b128((unsigned)(uintptr_t)&Bsl[buf][d * 40 + g * 8],
                    Wbase + (size_t)d * Ec + k0 + g * 8);
  };

  f32x8 acc[4] = {};

  stage_tile(0, 0);
  int buf = 0;
  for (int k0 = 0; k0 < Ec; k0 += 32, buf ^= 1) {
    if (k0 + 32 < Ec) {
      stage_tile(k0 + 32, buf ^ 1);
      asm volatile("s_wait_asynccnt 0x3" ::: "memory");  // current tile landed
    } else {
      asm volatile("s_wait_asynccnt 0x0" ::: "memory");
    }
    __syncthreads();  // all waves' copies for `buf` visible

    bf16x16 a = load_frag_row(&Asl[buf][(wave * 16 + l) * 40], hi);
    #pragma unroll
    for (int nt = 0; nt < 4; ++nt) {
      bf16x16 b = load_frag_row(&Bsl[buf][(nt * 16 + l) * 40], hi);
      acc[nt] = wmma_bf16(a, b, acc[nt]);
    }
    __syncthreads();  // all waves done reading `buf` before it is refilled
  }

  // epilogue: + bias, * scale, store bf16 (V goes out transposed)
  #pragma unroll
  for (int nt = 0; nt < 4; ++nt) {
    int n = nt * 16 + l;
    float bs = bias[h * DHc + n];
    #pragma unroll
    for (int r = 0; r < 8; ++r) {
      int mrow = m0 + wave * 16 + hi * 8 + r;  // accumulator C layout
      int bb = mrow / Tc, t = mrow % Tc;
      float val = (acc[nt][r] + bs) * scale;
      size_t oidx = (blockIdx.z == 2)
          ? (((size_t)bb * Hc + h) * DHc + n) * Tc + t    // V: [B,H,DH,T]
          : (((size_t)bb * Hc + h) * Tc + t) * DHc + n;   // Q,K: [B,H,T,DH]
      outp[oidx] = (__bf16)val;
    }
  }
}

// ---------------------------------------------------------------------------
// Kernel 2: causal flash attention. Q,K [B,H,T,DH]; V [B,H,DH,T] (transposed),
// so both QK^T and PV fragments load contiguously from L2-resident global.
// grid (B*H, T/64), block 128 (4 waves); no block barriers needed.
// ---------------------------------------------------------------------------
__global__ __launch_bounds__(128) void flash_attn_kernel(
    const __bf16* __restrict__ Q, const __bf16* __restrict__ K,
    const __bf16* __restrict__ V, float* __restrict__ out)
{
  __shared__ alignas(16) __bf16 Ps[4 * 16 * 72];  // per-wave P slab [row][key]

  const int bh = blockIdx.x;
  const int b  = bh / Hc, h = bh % Hc;
  const int q0 = blockIdx.y * 64;

  const int tid  = threadIdx.x;
  const int lane = tid & 31, wave = tid >> 5;
  const int l = lane & 15, hi = lane >> 4;

  const __bf16* Qb = Q + (size_t)(b * Hc + h) * Tc * DHc;
  const __bf16* Kb = K + (size_t)(b * Hc + h) * Tc * DHc;
  const __bf16* Vb = V + (size_t)(b * Hc + h) * DHc * Tc;  // [dh][t]

  // Q A-fragments for this wave's 16 rows (K-dim = dh, two halves)
  const __bf16* qp = Qb + (size_t)(q0 + wave * 16 + l) * DHc;
  bf16x16 qf0 = load_frag_row(qp, hi);
  bf16x16 qf1 = load_frag_row(qp + 32, hi);

  f32x8 o[4] = {};
  f32x8 mrow, lrow;
  #pragma unroll
  for (int r = 0; r < 8; ++r) { mrow[r] = -3.0e38f; lrow[r] = 0.0f; }

  __bf16* slab = &Ps[wave * (16 * 72)];

  for (int kb = 0; kb <= q0; kb += 64) {
    if (kb + 64 <= q0)  // prefetch next K tile
      __builtin_prefetch(Kb + (size_t)(kb + 64) * DHc, 0, 1);

    // S = Q * K^T  (B-frag column n = key row kb+n, contiguous over dh)
    f32x8 s[4] = {};
    #pragma unroll
    for (int nt = 0; nt < 4; ++nt) {
      const __bf16* kp = Kb + (size_t)(kb + nt * 16 + l) * DHc;
      s[nt] = wmma_bf16(qf0, load_frag_row(kp, hi), s[nt]);
      s[nt] = wmma_bf16(qf1, load_frag_row(kp + 32, hi), s[nt]);
    }

    // causal mask on the diagonal key tile
    if (kb == q0) {
      #pragma unroll
      for (int nt = 0; nt < 4; ++nt) {
        int key = kb + nt * 16 + l;
        #pragma unroll
        for (int r = 0; r < 8; ++r) {
          int qq = q0 + wave * 16 + hi * 8 + r;
          if (key > qq) s[nt][r] = -3.0e38f;
        }
      }
    }

    // online softmax: row max over 64 cols (16 lanes share a row)
    f32x8 mt;
    #pragma unroll
    for (int r = 0; r < 8; ++r)
      mt[r] = fmaxf(fmaxf(s[0][r], s[1][r]), fmaxf(s[2][r], s[3][r]));
    #pragma unroll
    for (int off = 1; off < 16; off <<= 1)
      #pragma unroll
      for (int r = 0; r < 8; ++r)
        mt[r] = fmaxf(mt[r], __shfl_xor(mt[r], off, 32));

    f32x8 esc, rsum;
    #pragma unroll
    for (int r = 0; r < 8; ++r) {
      float mn = fmaxf(mrow[r], mt[r]);
      esc[r] = __expf(mrow[r] - mn);
      mrow[r] = mn;
      rsum[r] = 0.0f;
    }
    #pragma unroll
    for (int nt = 0; nt < 4; ++nt)
      #pragma unroll
      for (int r = 0; r < 8; ++r) {
        float p = __expf(s[nt][r] - mrow[r]);
        s[nt][r] = p;
        rsum[r] += p;
      }
    #pragma unroll
    for (int off = 1; off < 16; off <<= 1)
      #pragma unroll
      for (int r = 0; r < 8; ++r)
        rsum[r] += __shfl_xor(rsum[r], off, 32);
    #pragma unroll
    for (int r = 0; r < 8; ++r) lrow[r] = lrow[r] * esc[r] + rsum[r];
    #pragma unroll
    for (int nt = 0; nt < 4; ++nt)
      #pragma unroll
      for (int r = 0; r < 8; ++r) o[nt][r] *= esc[r];

    // P -> bf16 A-fragment layout via per-wave LDS slab (wave-private: DS ops
    // from one wave stay in order, no block barrier required)
    #pragma unroll
    for (int nt = 0; nt < 4; ++nt)
      #pragma unroll
      for (int r = 0; r < 8; ++r)
        slab[(hi * 8 + r) * 72 + nt * 16 + l] = (__bf16)s[nt][r];

    // O += P * V   (K-dim = 64 keys, two halves; V rows contiguous over keys)
    #pragma unroll
    for (int kh = 0; kh < 2; ++kh) {
      bf16x16 a = load_frag_row(&slab[l * 72 + kh * 32], hi);
      #pragma unroll
      for (int nt = 0; nt < 4; ++nt) {
        const __bf16* vp = Vb + (size_t)(nt * 16 + l) * Tc + kb + kh * 32;
        o[nt] = wmma_bf16(a, load_frag_row(vp, hi), o[nt]);
      }
    }
  }

  // epilogue: /l, round to 5 decimals, scatter to [B,T,H*DH] f32
  #pragma unroll
  for (int nt = 0; nt < 4; ++nt)
    #pragma unroll
    for (int r = 0; r < 8; ++r) {
      float val = o[nt][r] / lrow[r];
      val = rintf(val * 1.0e5f) * 1.0e-5f;
      int t  = q0 + wave * 16 + hi * 8 + r;
      int dh = nt * 16 + l;
      out[((size_t)b * Tc + t) * (Hc * DHc) + h * DHc + dh] = val;
    }
}

// ---------------------------------------------------------------------------
extern "C" void kernel_launch(void* const* d_in, const int* in_sizes, int n_in,
                              void* d_out, int out_size, void* d_ws, size_t ws_size,
                              hipStream_t stream) {
  (void)in_sizes; (void)n_in; (void)out_size; (void)ws_size;
  const float* embed = (const float*)d_in[0];
  const float* Wq = (const float*)d_in[1];
  const float* bq = (const float*)d_in[2];
  const float* Wk = (const float*)d_in[3];
  const float* bk = (const float*)d_in[4];
  const float* Wv = (const float*)d_in[5];
  const float* bv = (const float*)d_in[6];
  float* out = (float*)d_out;

  const size_t e_elems = (size_t)BT * Ec;            // 4 Mi
  const size_t w_elems = (size_t)Hc * Ec * DHc;      // 1 Mi
  const size_t qkv_elems = (size_t)Bc * Hc * Tc * DHc;  // 4 Mi
  __bf16* Ebf = (__bf16*)d_ws;
  __bf16* Wtq = Ebf + e_elems;
  __bf16* Wtk = Wtq + w_elems;
  __bf16* Wtv = Wtk + w_elems;
  __bf16* Qw  = Wtv + w_elems;
  __bf16* Kw  = Qw + qkv_elems;
  __bf16* Vw  = Kw + qkv_elems;

  convert_embed_kernel<<<dim3(e_elems / (256 * 8)), 256, 0, stream>>>(embed, Ebf);
  convert_weights_kernel<<<dim3(Ec / 32, Hc, 3), 256, 0, stream>>>(
      Wq, Wk, Wv, Wtq, Wtk, Wtv);

  qkv_proj_kernel<<<dim3(BT / 128, Hc, 3), 256, 0, stream>>>(
      Ebf, Wtq, bq, Wtk, bk, Wtv, bv, Qw, Kw, Vw);

  flash_attn_kernel<<<dim3(Bc * Hc, Tc / 64), 128, 0, stream>>>(Qw, Kw, Vw, out);
}
